// LayerRouter_63479616635484
// MI455X (gfx1250) — matrix operand order repaired and checked by reading, not compiled, not run on hardware
//
#include <hip/hip_runtime.h>
#include <hip/hip_bf16.h>

// ---------------- problem sizes (from reference) ----------------
#define T_TOKENS   16384      // B*S = 4*4096
#define H_DIM      4096
#define MID_DIM    2048
#define NEXP       8

// ---------------- tiling ----------------
#define BM      64            // token rows per block (4 M-tiles of 16)
#define BN      128           // MID columns per n-chunk (8 N-tiles of 16)
#define KT      32            // K per bf16 WMMA
#define NCHUNK  (MID_DIM / BN)   // 16
#define STR     48            // LDS row stride in bf16 elems (96B, 16B-aligned)
#define THREADS 256           // 8 waves (wave32)
#define TT      64            // transpose tile

typedef __bf16 bf16_t;
typedef __attribute__((ext_vector_type(16))) __bf16 v16bf;
typedef __attribute__((ext_vector_type(8)))  __bf16 v8bf;
typedef __attribute__((ext_vector_type(8)))  float  v8f;

union FragBF {
  v16bf v;
  v8bf  h[2];
};

__device__ __forceinline__ unsigned int bf16_rne_bits(float f) {
  unsigned int u = __float_as_uint(f);
  return u + 0x7FFFu + ((u >> 16) & 1u);   // round-to-nearest-even (pre-shift)
}
// f = hi(bf16) + lo(f32 residual); hbits = hi as 16-bit pattern
__device__ __forceinline__ void split1(float f, unsigned int& hbits, float& lo) {
  unsigned int r = bf16_rne_bits(f);
  lo = f - __uint_as_float(r & 0xFFFF0000u);
  hbits = r >> 16;
}

// CDNA5 async global->LDS copy (16B per lane), tracked by ASYNCcnt
__device__ __forceinline__ void async_cp16(const void* gsrc, void* ldst) {
  unsigned lds = (unsigned)(unsigned long long)ldst;  // low 32b of generic = LDS offset
  asm volatile("global_load_async_to_lds_b128 %0, %1, off"
               :: "v"(lds), "v"(gsrc)
               : "memory");
}
__device__ __forceinline__ void wait_async0() {
  asm volatile("s_wait_asynccnt 0x0" ::: "memory");
}

// -------- one-time pass 1: x (f32) -> hi/lo bf16 planes, same layout --------
__global__ __launch_bounds__(256) void split_x_kernel(
    const float4* __restrict__ src, uint2* __restrict__ hi,
    uint2* __restrict__ lo, long n4)
{
  long i = (long)blockIdx.x * blockDim.x + threadIdx.x;
  const long stride = (long)gridDim.x * blockDim.x;
  for (; i < n4; i += stride) {
    float4 f = src[i];
    float v[4] = {f.x, f.y, f.z, f.w};
    unsigned int hb[4], lb[4];
#pragma unroll
    for (int j = 0; j < 4; ++j) {
      float lv;
      split1(v[j], hb[j], lv);
      lb[j] = bf16_rne_bits(lv) >> 16;
    }
    hi[i] = make_uint2(hb[0] | (hb[1] << 16), hb[2] | (hb[3] << 16));
    lo[i] = make_uint2(lb[0] | (lb[1] << 16), lb[2] | (lb[3] << 16));
  }
}

// -------- one-time pass 2: w1 [k][n] f32 -> transposed [n][k] bf16 planes ---
__global__ __launch_bounds__(256) void split_w1t_kernel(
    const float* __restrict__ w1, bf16_t* __restrict__ hi,
    bf16_t* __restrict__ lo)
{
  __shared__ bf16_t th[TT][TT + 2];
  __shared__ bf16_t tl[TT][TT + 2];
  const int t  = threadIdx.x;
  const int k0 = blockIdx.x * TT;
  const int n0 = blockIdx.y * TT;
  {
    const int nl = t & 63, kl0 = t >> 6;
#pragma unroll
    for (int p = 0; p < 16; ++p) {
      int kl = kl0 + p * 4;
      float f = w1[(size_t)(k0 + kl) * MID_DIM + (n0 + nl)];
      unsigned int hb; float lv;
      split1(f, hb, lv);
      th[kl][nl] = __builtin_bit_cast(bf16_t, (unsigned short)hb);
      tl[kl][nl] = __builtin_bit_cast(bf16_t, (unsigned short)(bf16_rne_bits(lv) >> 16));
    }
  }
  __syncthreads();
  {
    const int kl = t & 63, nl0 = t >> 6;
#pragma unroll
    for (int p = 0; p < 16; ++p) {
      int n = nl0 + p * 4;
      hi[(size_t)(n0 + n) * H_DIM + k0 + kl] = th[kl][n];
      lo[(size_t)(n0 + n) * H_DIM + k0 + kl] = tl[kl][n];
    }
  }
}

// -------- fused: relu(x@w1+b1) via split-bf16 WMMA -> @w2 -> softmax/top2 ---
__global__ __launch_bounds__(THREADS) void moe_router_fused(
    const bf16_t* __restrict__ xh,  const bf16_t* __restrict__ xl,
    const bf16_t* __restrict__ w1th, const bf16_t* __restrict__ w1tl,
    const float* __restrict__ b1, const float* __restrict__ w2,
    const float* __restrict__ b2, float* __restrict__ mask_out,
    float* __restrict__ red_ws)
{
  // double-buffered operand tiles: [buf][hi/lo][...]
  __shared__ __align__(16) bf16_t sA[2][2][BM * STR];
  __shared__ __align__(16) bf16_t sB[2][2][BN * STR];
  __shared__ float sLogit[4 * 16 * NEXP];   // [waveM][row in tile][expert]
  __shared__ float sRed[2 * NEXP];          // usage[8], mass[8]

  const int t     = threadIdx.x;
  const int wave  = t >> 5;
  const int lane  = t & 31;
  const int l15   = lane & 15;
  const int half  = lane >> 4;
  const int waveM = wave >> 1;              // 16-row tile (0..3)
  const int waveN = wave & 1;               // which 4 of the 8 col tiles
  const int rowBase = blockIdx.x * BM;

  for (int i = t; i < 4 * 16 * NEXP; i += THREADS) sLogit[i] = 0.f;
  if (t < 2 * NEXP) sRed[t] = 0.f;

  float partial[8][NEXP];
#pragma unroll
  for (int i = 0; i < 8; ++i)
#pragma unroll
    for (int e = 0; e < NEXP; ++e) partial[i][e] = 0.f;

  // staging coordinates: async b128 copies, no VGPR round trip
  const int aRow = t >> 2, aKg = (t & 3) * 8;   // A: [64 x 32] per plane, 16B/thread
  const int bRow = t >> 1, bKg = (t & 1) * 16;  // B: [128 x 32] per plane, 32B/thread
  const size_t aSrc = (size_t)(rowBase + aRow) * H_DIM + aKg;

  // WMMA fragment coordinates (ISA 7.12.2 layouts)
  const int arow = waveM * 16 + l15;
  const int kA = half ? 8 : 0;              // A lane: K {0-7,16-23} / {8-15,24-31}
  const int kB = half ? 8 : 0;              // B lane: K {0-7,16-23} / {8-15,24-31}

  for (int nc = 0; nc < NCHUNK; ++nc) {
    const int n0 = nc * BN;
    const size_t bSrc = (size_t)(n0 + bRow) * H_DIM + bKg;

    auto stage = [&](int kk, int b) {
      async_cp16(&xh[aSrc + kk],       &sA[b][0][aRow * STR + aKg]);
      async_cp16(&xl[aSrc + kk],       &sA[b][1][aRow * STR + aKg]);
      async_cp16(&w1th[bSrc + kk],     &sB[b][0][bRow * STR + bKg]);
      async_cp16(&w1th[bSrc + kk + 8], &sB[b][0][bRow * STR + bKg + 8]);
      async_cp16(&w1tl[bSrc + kk],     &sB[b][1][bRow * STR + bKg]);
      async_cp16(&w1tl[bSrc + kk + 8], &sB[b][1][bRow * STR + bKg + 8]);
    };

    v8f acc[4];
#pragma unroll
    for (int c = 0; c < 4; ++c)
#pragma unroll
      for (int i = 0; i < 8; ++i) acc[c][i] = 0.f;

    stage(0, 0);                          // prologue: tile k=0 -> buffer 0
    int buf = 0;

    for (int kk = 0; kk < H_DIM; kk += KT) {
      wait_async0();                      // this wave's copies landed in LDS
      __syncthreads();                    // all waves' copies landed
      if (kk + KT < H_DIM) stage(kk + KT, buf ^ 1);  // overlap next fetch

      const bf16_t* Ah = sA[buf][0];
      const bf16_t* Al = sA[buf][1];
      const bf16_t* Bh = sB[buf][0];
      const bf16_t* Bl = sB[buf][1];

      FragBF ahi, alo;
      ahi.h[0] = *(const v8bf*)&Ah[arow * STR + kA];
      ahi.h[1] = *(const v8bf*)&Ah[arow * STR + kA + 16];
      alo.h[0] = *(const v8bf*)&Al[arow * STR + kA];
      alo.h[1] = *(const v8bf*)&Al[arow * STR + kA + 16];

#pragma unroll
      for (int c = 0; c < 4; ++c) {
        int bn = (waveN * 4 + c) * 16 + l15;
        FragBF bhi, blo;
        bhi.h[0] = *(const v8bf*)&Bh[bn * STR + kB];
        bhi.h[1] = *(const v8bf*)&Bh[bn * STR + kB + 16];
        blo.h[0] = *(const v8bf*)&Bl[bn * STR + kB];
        blo.h[1] = *(const v8bf*)&Bl[bn * STR + kB + 16];
        // split-bf16 fp32 emulation: hi*hi + hi*lo + lo*hi
        acc[c] = __builtin_amdgcn_wmma_f32_16x16x32_bf16(
            false, ahi.v, false, bhi.v, (short)0, acc[c], false, false);
        acc[c] = __builtin_amdgcn_wmma_f32_16x16x32_bf16(
            false, ahi.v, false, blo.v, (short)0, acc[c], false, false);
        acc[c] = __builtin_amdgcn_wmma_f32_16x16x32_bf16(
            false, alo.v, false, bhi.v, (short)0, acc[c], false, false);
      }
      buf ^= 1;
    }

    // chunk epilogue: h = relu(acc + b1), fold into logits via w2 (GEMM2)
#pragma unroll
    for (int c = 0; c < 4; ++c) {
      int gcol = n0 + (waveN * 4 + c) * 16 + l15;
      float b1v = b1[gcol];
      const float4* w2p = (const float4*)(w2 + (size_t)gcol * NEXP);
      float4 wlo = w2p[0], whi = w2p[1];
      float we[NEXP] = {wlo.x, wlo.y, wlo.z, wlo.w, whi.x, whi.y, whi.z, whi.w};
#pragma unroll
      for (int i = 0; i < 8; ++i) {
        float hv = fmaxf(acc[c][i] + b1v, 0.f);
#pragma unroll
        for (int e = 0; e < NEXP; ++e)
          partial[i][e] = fmaf(hv, we[e], partial[i][e]);
      }
    }
  }

  // cross-lane/column reduction of logit partials (C layout: lane=N, vgpr=M)
#pragma unroll
  for (int i = 0; i < 8; ++i)
#pragma unroll
    for (int e = 0; e < NEXP; ++e)
      atomicAdd(&sLogit[waveM * 128 + (half * 8 + i) * NEXP + e], partial[i][e]);
  __syncthreads();

  // per-token router: softmax over 8 experts, top-2, normalized soft mask
  if (t < BM) {
    float lg[NEXP];
#pragma unroll
    for (int e = 0; e < NEXP; ++e)
      lg[e] = sLogit[(t >> 4) * 128 + (t & 15) * NEXP + e] + b2[e];
    float m = lg[0];
#pragma unroll
    for (int e = 1; e < NEXP; ++e) m = fmaxf(m, lg[e]);
    float p[NEXP], s = 0.f;
#pragma unroll
    for (int e = 0; e < NEXP; ++e) { p[e] = __expf(lg[e] - m); s += p[e]; }
    float inv = 1.f / s;
#pragma unroll
    for (int e = 0; e < NEXP; ++e) p[e] *= inv;
    int i1 = 0; float p1 = p[0];
#pragma unroll
    for (int e = 1; e < NEXP; ++e) if (p[e] > p1) { p1 = p[e]; i1 = e; }
    int i2 = -1; float p2 = -1.f;
#pragma unroll
    for (int e = 0; e < NEXP; ++e) if (e != i1 && p[e] > p2) { p2 = p[e]; i2 = e; }
    float denom = fmaxf(p1 + p2, 1e-8f);
    float m1 = p1 / denom, m2 = p2 / denom;
    float ov[NEXP];
#pragma unroll
    for (int e = 0; e < NEXP; ++e)
      ov[e] = (e == i1) ? m1 : ((e == i2) ? m2 : 0.f);
    size_t row = (size_t)rowBase + t;
    float4* op = (float4*)(mask_out + row * NEXP);
    op[0] = make_float4(ov[0], ov[1], ov[2], ov[3]);
    op[1] = make_float4(ov[4], ov[5], ov[6], ov[7]);
    atomicAdd(&sRed[i1], 1.f);
    atomicAdd(&sRed[i2], 1.f);
    atomicAdd(&sRed[NEXP + i1], m1);
    atomicAdd(&sRed[NEXP + i2], m2);
  }
  __syncthreads();
  if (t < 2 * NEXP) red_ws[blockIdx.x * (2 * NEXP) + t] = sRed[t];
}

// aux_loss = E * sum_e mean(usage_e) * mean(mass_e); deterministic final pass
__global__ void moe_aux_kernel(const float* __restrict__ red_ws,
                               float* __restrict__ out_aux, int nblocks)
{
  if (threadIdx.x == 0 && blockIdx.x == 0) {
    float usage[NEXP], mass[NEXP];
#pragma unroll
    for (int e = 0; e < NEXP; ++e) { usage[e] = 0.f; mass[e] = 0.f; }
    for (int b = 0; b < nblocks; ++b)
#pragma unroll
      for (int e = 0; e < NEXP; ++e) {
        usage[e] += red_ws[b * (2 * NEXP) + e];
        mass[e]  += red_ws[b * (2 * NEXP) + NEXP + e];
      }
    const float invT = 1.0f / (float)T_TOKENS;
    float s = 0.f;
#pragma unroll
    for (int e = 0; e < NEXP; ++e) s += (usage[e] * invT) * (mass[e] * invT);
    out_aux[0] = (float)NEXP * s;
  }
}

extern "C" void kernel_launch(void* const* d_in, const int* in_sizes, int n_in,
                              void* d_out, int out_size, void* d_ws, size_t ws_size,
                              hipStream_t stream) {
  const float* x  = (const float*)d_in[0];
  const float* w1 = (const float*)d_in[1];
  const float* b1 = (const float*)d_in[2];
  const float* w2 = (const float*)d_in[3];
  const float* b2 = (const float*)d_in[4];
  float* out = (float*)d_out;

  // workspace layout
  char* wsb = (char*)d_ws;
  float*  red  = (float*)wsb;                                   // 16 KB
  bf16_t* xh   = (bf16_t*)(wsb + (1 << 16));                    // 134 MB
  bf16_t* xl   = xh + (size_t)T_TOKENS * H_DIM;                 // 134 MB
  bf16_t* w1th = xl + (size_t)T_TOKENS * H_DIM;                 // 16.8 MB
  bf16_t* w1tl = w1th + (size_t)H_DIM * MID_DIM;                // 16.8 MB

  // pass 1: split x into bf16 hi/lo planes (one-time, HBM-bound)
  const long n4 = (long)T_TOKENS * H_DIM / 4;
  split_x_kernel<<<8192, 256, 0, stream>>>(
      (const float4*)x, (uint2*)xh, (uint2*)xl, n4);

  // pass 2: transpose+split w1 -> [n][k] bf16 hi/lo planes
  dim3 tgrid(H_DIM / TT, MID_DIM / TT);
  split_w1t_kernel<<<tgrid, 256, 0, stream>>>(w1, w1th, w1tl);

  // pass 3: fused GEMM1 (split-bf16 WMMA) + GEMM2 + router epilogue
  const int nblocks = T_TOKENS / BM;   // 256 blocks, 8 wave32 waves each
  moe_router_fused<<<nblocks, THREADS, 0, stream>>>(
      xh, xl, w1th, w1tl, b1, w2, b2, out, red);

  // pass 4: deterministic aux-loss reduction
  moe_aux_kernel<<<1, 32, 0, stream>>>(red, out + (size_t)T_TOKENS * NEXP, nblocks);

  (void)in_sizes; (void)n_in; (void)out_size; (void)ws_size;
}
